// Transformer_87479893885684
// MI455X (gfx1250) — matrix operand order repaired and checked by reading, not compiled, not run on hardware
//
#include <hip/hip_runtime.h>

// Problem constants (match reference).
#define S_  1024
#define D_  768
#define H_  12
#define L_  6
#define V_  32000
#define KD_ 64
#define FF_ 3072
#define NB_ 2
#define PAD_ 0

typedef __attribute__((ext_vector_type(16))) __bf16 v16bf;
typedef __attribute__((ext_vector_type(8)))  float  v8f;

__device__ __forceinline__ unsigned f2bf(float f) {
  unsigned u = __float_as_uint(f);
  u += 0x7fffu + ((u >> 16) & 1u);   // round-to-nearest-even truncation to bf16
  return u >> 16;
}

union FragBF { uint4 q[2]; v16bf v; };

// ---------------------------------------------------------------------------
// Tiled bf16 WMMA GEMM:  C[bz] = alpha * A[bz] x B[bz] + bias[bz]  (opt relu /
// atomic accumulate).  A:[M,K] row-major f32, B:[K,N] row-major f32 (or, with
// TRANS_B, stored as [N,K] and read transposed), C:[M,N] f32.
//
// BK = 32 = the WMMA K, so each LDS stage feeds exactly one wmma K-slice.
// Software pipeline: tile k+1 is loaded from global and converted to packed
// bf16 in registers while the WMMAs for tile k run out of LDS buffer (k&1);
// the prefetched tile is then stored into the other LDS buffer and a single
// workgroup barrier per K-step publishes it.  global_prefetch warms L2 two
// tiles ahead.
// ---------------------------------------------------------------------------
template<int BM, int BN, int WM, int WN, bool TRANS_B>
__global__ __launch_bounds__(256) void gemm_wmma(
    const float* __restrict__ A, long a_bs, int lda,
    const float* __restrict__ Bm, long b_bs, int ldb,
    float* __restrict__ C, long c_bs, int ldc,
    const float* __restrict__ bias, long bias_bs,
    int K, float alpha, int relu, int atomic)
{
  constexpr int BK = 32;
  constexpr int MT = BM / (16 * WM);
  constexpr int NT = BN / (16 * WN);
  constexpr int SA = BK + 8;     // padded LDS row strides (16B-aligned rows,
  constexpr int SB = BN + 8;     // bank spread)
  constexpr int A_IT = BM * 8 / 256;        // float4 loads per thread (A)
  constexpr int B_IT = BK * BN / 4 / 256;   // float4 loads per thread (B)

  __shared__ unsigned short sA[2][BM * SA];
  __shared__ unsigned short sB[2][BK * SB];

  const int tid  = threadIdx.x;
  const int lane = tid & 31;
  const int wave = tid >> 5;
  const int half = lane >> 4;
  const int l16  = lane & 15;
  const int wm   = wave / WN;
  const int wn   = wave % WN;

  const float* Ab = A  + (long)blockIdx.z * a_bs;
  const float* Bb = Bm + (long)blockIdx.z * b_bs;
  float*       Cb = C  + (long)blockIdx.z * c_bs;
  const int m0 = blockIdx.y * BM;
  const int n0 = blockIdx.x * BN;

  // Register-resident prefetch staging (already bf16-packed for A / dense B).
  uint2 pa[A_IT];
  uint2 pb[B_IT];
  unsigned short tb[B_IT][4];

  auto load_tiles = [&](int k0) {
#pragma unroll
    for (int i = 0; i < A_IT; i++) {
      int idx = tid + i * 256;
      int r = idx >> 3, c = idx & 7;                  // 8 float4 per A row
      float4 f = *(const float4*)(Ab + (long)(m0 + r) * lda + (k0 + c * 4));
      pa[i].x = f2bf(f.x) | (f2bf(f.y) << 16);
      pa[i].y = f2bf(f.z) | (f2bf(f.w) << 16);
    }
    if (!TRANS_B) {
#pragma unroll
      for (int i = 0; i < B_IT; i++) {
        int idx = tid + i * 256;
        int r = idx / (BN / 4), c = idx % (BN / 4);
        float4 f = *(const float4*)(Bb + (long)(k0 + r) * ldb + (n0 + c * 4));
        pb[i].x = f2bf(f.x) | (f2bf(f.y) << 16);
        pb[i].y = f2bf(f.z) | (f2bf(f.w) << 16);
      }
    } else {
      // B(k,n) = Bb[(n0+n)*ldb + k0+k]  (e.g. K^T for Q.K^T)
#pragma unroll
      for (int i = 0; i < B_IT; i++) {
        int idx = tid + i * 256;
        int n = idx >> 3, c = idx & 7;                // 8 float4 along K per n
        float4 f = *(const float4*)(Bb + (long)(n0 + n) * ldb + (k0 + c * 4));
        tb[i][0] = (unsigned short)f2bf(f.x);
        tb[i][1] = (unsigned short)f2bf(f.y);
        tb[i][2] = (unsigned short)f2bf(f.z);
        tb[i][3] = (unsigned short)f2bf(f.w);
      }
    }
  };

  auto store_tiles = [&](int buf) {
#pragma unroll
    for (int i = 0; i < A_IT; i++) {
      int idx = tid + i * 256;
      int r = idx >> 3, c = idx & 7;
      *(uint2*)&sA[buf][r * SA + c * 4] = pa[i];      // ds_store_b64
    }
    if (!TRANS_B) {
#pragma unroll
      for (int i = 0; i < B_IT; i++) {
        int idx = tid + i * 256;
        int r = idx / (BN / 4), c = idx % (BN / 4);
        *(uint2*)&sB[buf][r * SB + c * 4] = pb[i];
      }
    } else {
#pragma unroll
      for (int i = 0; i < B_IT; i++) {
        int idx = tid + i * 256;
        int n = idx >> 3, c = idx & 7;
        int kk = c * 4;
        sB[buf][(kk + 0) * SB + n] = tb[i][0];
        sB[buf][(kk + 1) * SB + n] = tb[i][1];
        sB[buf][(kk + 2) * SB + n] = tb[i][2];
        sB[buf][(kk + 3) * SB + n] = tb[i][3];
      }
    }
  };

  v8f acc[MT][NT];
#pragma unroll
  for (int i = 0; i < MT; i++)
#pragma unroll
    for (int j = 0; j < NT; j++)
#pragma unroll
      for (int e = 0; e < 8; e++) acc[i][j][e] = 0.0f;

  const int nk = K / BK;

  // Prologue: stage tile 0.
  load_tiles(0);
  store_tiles(0);
  __syncthreads();

  for (int kt = 0; kt < nk; kt++) {
    const int cur = kt & 1;
    const bool more = (kt + 1 < nk);

    // Register prefetch of tile k+1 (overlaps with WMMA below).
    if (more) load_tiles((kt + 1) * BK);

    // Warm L2 two tiles ahead (global_prefetch_b8).
    if (kt + 2 < nk) {
      __builtin_prefetch(Ab + (long)(m0 + (tid >> 3)) * lda + (kt + 2) * BK, 0, 1);
      if (!TRANS_B)
        __builtin_prefetch(Bb + (long)((kt + 2) * BK + (tid >> 3)) * ldb + n0, 0, 1);
      else
        __builtin_prefetch(Bb + (long)(n0 + (tid >> 3)) * ldb + (kt + 2) * BK, 0, 1);
    }

    // ---- fragments per ISA 7.12.2 layouts + WMMA ------------------------
    FragBF af[MT], bfr[NT];
#pragma unroll
    for (int mt = 0; mt < MT; mt++) {
      // A lane: M = l16, holds K = {8*half..8*half+7, 16+8*half..16+8*half+7}
      const uint4* rowA =
          (const uint4*)&sA[cur][((wm * MT + mt) * 16 + l16) * SA];
      af[mt].q[0] = rowA[half];
      af[mt].q[1] = rowA[half + 2];
    }
#pragma unroll
    for (int nt = 0; nt < NT; nt++) {
      // B lane: N = l16, holds K = {8*half..8*half+7, 16+8*half..16+8*half+7}
      const unsigned short* colB = &sB[cur][(wn * NT + nt) * 16 + l16];
      unsigned w[8];
#pragma unroll
      for (int j = 0; j < 4; j++) {
        w[j]     = (unsigned)colB[(8 * half + 2 * j) * SB] | ((unsigned)colB[(8 * half + 2 * j + 1) * SB] << 16);
        w[4 + j] = (unsigned)colB[(16 + 8 * half + 2 * j) * SB] | ((unsigned)colB[(16 + 8 * half + 2 * j + 1) * SB] << 16);
      }
      bfr[nt].q[0] = make_uint4(w[0], w[1], w[2], w[3]);
      bfr[nt].q[1] = make_uint4(w[4], w[5], w[6], w[7]);
    }
#pragma unroll
    for (int mt = 0; mt < MT; mt++)
#pragma unroll
      for (int nt = 0; nt < NT; nt++)
        acc[mt][nt] = __builtin_amdgcn_wmma_f32_16x16x32_bf16(
            false, af[mt].v, false, bfr[nt].v, (short)0, acc[mt][nt],
            false, false);

    // Publish tile k+1 into the other buffer; one barrier per K-step.
    if (more) store_tiles(cur ^ 1);
    __syncthreads();
  }

  // ---- epilogue: alpha, bias, relu, plain store or atomic accumulate ----
#pragma unroll
  for (int mt = 0; mt < MT; mt++) {
    int mbase = m0 + (wm * MT + mt) * 16 + half * 8;   // C: M = vgpr + 8*half
#pragma unroll
    for (int nt = 0; nt < NT; nt++) {
      int nn = n0 + (wn * NT + nt) * 16 + l16;         // C: N = lane%16
      float bv = bias ? bias[bias_bs * blockIdx.z + nn] : 0.0f;
#pragma unroll
      for (int e = 0; e < 8; e++) {
        float val = acc[mt][nt][e] * alpha + bv;
        if (relu) val = fmaxf(val, 0.0f);
        float* p = Cb + (long)(mbase + e) * ldc + nn;
        if (atomic) atomicAdd(p, val);
        else        *p = val;
      }
    }
  }
}

// ---------------------------------------------------------------------------
// Embedding + sinusoidal positional encoding:  x[b,s,d] = emb[tok[b,s],d]+pe
// ---------------------------------------------------------------------------
__global__ __launch_bounds__(256) void k_embed(const int* __restrict__ tok,
                                               const float* __restrict__ emb,
                                               float* __restrict__ x)
{
  long i = (long)blockIdx.x * 256 + threadIdx.x;       // over B*S*D
  int d   = (int)(i % D_);
  long bs = i / D_;
  int s   = (int)(bs % S_);
  int t   = tok[bs];
  int j   = d >> 1;
  float ang = (float)s * __expf((float)(2 * j) * (-9.210340372f / (float)D_));
  float pe  = (d & 1) ? __cosf(ang) : __sinf(ang);
  x[i] = emb[(long)t * D_ + d] + pe;
}

__global__ __launch_bounds__(256) void k_zero(float* __restrict__ p)
{
  p[(long)blockIdx.x * 256 + threadIdx.x] = 0.0f;
}

// ---------------------------------------------------------------------------
// Masked softmax over one attention row.  mask = causal | pad-key | pad-query;
// fully-masked rows -> uniform 1/S (reference semantics).  scale already
// applied by the scores GEMM.  grid = (S, H), sc = scores for this batch elem.
// ---------------------------------------------------------------------------
__global__ __launch_bounds__(256) void k_softmax(float* __restrict__ sc,
                                                 const int* __restrict__ tok)
{
  const int srow = blockIdx.x;
  const int h    = blockIdx.y;
  float* row = sc + ((long)h * S_ + srow) * (long)S_;
  const bool padq = (tok[srow] == PAD_);
  const int tid = threadIdx.x;
  __shared__ float red[256];

  float vals[S_ / 256];
  bool  msk[S_ / 256];
  float mx = -3.0e38f;
#pragma unroll
  for (int i = 0; i < S_ / 256; i++) {
    int t = tid + i * 256;
    bool m = padq || (t > srow) || (tok[t] == PAD_);
    msk[i]  = m;
    vals[i] = row[t];
    if (!m) mx = fmaxf(mx, vals[i]);
  }
  red[tid] = mx;
  __syncthreads();
  for (int st = 128; st > 0; st >>= 1) {
    if (tid < st) red[tid] = fmaxf(red[tid], red[tid + st]);
    __syncthreads();
  }
  const float mxv = red[0];
  __syncthreads();
  const bool uniform = (mxv < -2.9e38f);

  float e[S_ / 256];
  float sum = 0.0f;
#pragma unroll
  for (int i = 0; i < S_ / 256; i++) {
    e[i] = (msk[i] || uniform) ? 0.0f : __expf(vals[i] - mxv);
    sum += e[i];
  }
  red[tid] = sum;
  __syncthreads();
  for (int st = 128; st > 0; st >>= 1) {
    if (tid < st) red[tid] += red[tid + st];
    __syncthreads();
  }
  const float inv = uniform ? (1.0f / (float)S_) : (1.0f / red[0]);
#pragma unroll
  for (int i = 0; i < S_ / 256; i++)
    row[tid + i * 256] = uniform ? inv : e[i] * inv;
}

// ---------------------------------------------------------------------------
// Residual add + LayerNorm over a D=768 row; writes back into x (in place).
// ---------------------------------------------------------------------------
__global__ __launch_bounds__(256) void k_add_ln(float* __restrict__ x,
                                                const float* __restrict__ add,
                                                const float* __restrict__ g,
                                                const float* __restrict__ be)
{
  const int tid = threadIdx.x;
  float* row        = x   + (long)blockIdx.x * D_;
  const float* arow = add + (long)blockIdx.x * D_;
  float y[D_ / 256];
  float s = 0.0f, s2 = 0.0f;
#pragma unroll
  for (int i = 0; i < D_ / 256; i++) {
    int d = tid + i * 256;
    y[i] = row[d] + arow[d];
    s  += y[i];
    s2 += y[i] * y[i];
  }
  __shared__ float r1[256], r2[256];
  r1[tid] = s; r2[tid] = s2;
  __syncthreads();
  for (int st = 128; st > 0; st >>= 1) {
    if (tid < st) { r1[tid] += r1[tid + st]; r2[tid] += r2[tid + st]; }
    __syncthreads();
  }
  const float m   = r1[0] / (float)D_;
  const float var = r2[0] / (float)D_ - m * m;
  const float inv = rsqrtf(var + 1e-5f);
#pragma unroll
  for (int i = 0; i < D_ / 256; i++) {
    int d = tid + i * 256;
    row[d] = (y[i] - m) * inv * g[d] + be[d];
  }
}

// ---------------------------------------------------------------------------
extern "C" void kernel_launch(void* const* d_in, const int* in_sizes, int n_in,
                              void* d_out, int out_size, void* d_ws, size_t ws_size,
                              hipStream_t stream)
{
  (void)in_sizes; (void)n_in; (void)out_size; (void)ws_size;
  const int*   tok  = (const int*)d_in[0];
  const float* emb  = (const float*)d_in[1];
  const float* Wq   = (const float*)d_in[2];
  const float* bq   = (const float*)d_in[3];
  const float* Wk   = (const float*)d_in[4];
  const float* bk   = (const float*)d_in[5];
  const float* Wv   = (const float*)d_in[6];
  const float* bv   = (const float*)d_in[7];
  const float* g1   = (const float*)d_in[8];
  const float* be1  = (const float*)d_in[9];
  const float* g2   = (const float*)d_in[10];
  const float* be2  = (const float*)d_in[11];
  const float* W1   = (const float*)d_in[12];
  const float* b1   = (const float*)d_in[13];
  const float* W2   = (const float*)d_in[14];
  const float* b2   = (const float*)d_in[15];
  const float* Wout = (const float*)d_in[16];
  const float* bout = (const float*)d_in[17];
  float* out = (float*)d_out;

  // Workspace carve (f32): ~120 MB total, per-batch attention buffers reused.
  float* ws    = (float*)d_ws;
  float* x     = ws;  ws += (long)NB_ * S_ * D_;   // persistent activations
  float* q     = ws;  ws += (long)H_ * S_ * KD_;
  float* kbuf  = ws;  ws += (long)H_ * S_ * KD_;
  float* v     = ws;  ws += (long)H_ * S_ * D_;
  float* sc    = ws;  ws += (long)H_ * S_ * S_;
  float* tmp   = ws;  ws += (long)S_ * D_;         // head-avg / ffn out
  float* ffh   = ws;  ws += (long)S_ * FF_;

  k_embed<<<dim3(NB_ * S_ * D_ / 256), dim3(256), 0, stream>>>(tok, emb, x);

  for (int l = 0; l < L_; l++) {
    for (int b = 0; b < NB_; b++) {
      float* xb = x + (long)b * S_ * D_;
      const int* tokb = tok + b * S_;

      // Q = x Wq[l,h] + bq   (per head, N=64)
      gemm_wmma<128, 64, 4, 2, false><<<dim3(1, S_ / 128, H_), 256, 0, stream>>>(
          xb, 0L, D_,
          Wq + (long)l * H_ * D_ * KD_, (long)D_ * KD_, KD_,
          q, (long)S_ * KD_, KD_,
          bq + (long)l * H_ * KD_, (long)KD_,
          D_, 1.0f, 0, 0);
      // K = x Wk[l,h] + bk
      gemm_wmma<128, 64, 4, 2, false><<<dim3(1, S_ / 128, H_), 256, 0, stream>>>(
          xb, 0L, D_,
          Wk + (long)l * H_ * D_ * KD_, (long)D_ * KD_, KD_,
          kbuf, (long)S_ * KD_, KD_,
          bk + (long)l * H_ * KD_, (long)KD_,
          D_, 1.0f, 0, 0);
      // V = x Wv[l,h] + bv   (per head, N=768)
      gemm_wmma<128, 128, 2, 4, false><<<dim3(D_ / 128, S_ / 128, H_), 256, 0, stream>>>(
          xb, 0L, D_,
          Wv + (long)l * H_ * D_ * D_, (long)D_ * D_, D_,
          v, (long)S_ * D_, D_,
          bv + (long)l * H_ * D_, (long)D_,
          D_, 1.0f, 0, 0);
      // scores = (Q K^T) / sqrt(KD)
      gemm_wmma<128, 128, 2, 4, true><<<dim3(S_ / 128, S_ / 128, H_), 256, 0, stream>>>(
          q, (long)S_ * KD_, KD_,
          kbuf, (long)S_ * KD_, KD_,
          sc, (long)S_ * S_, S_,
          nullptr, 0L,
          KD_, 0.125f, 0, 0);
      // masked softmax (causal | pad-key | pad-query; all-masked -> uniform)
      k_softmax<<<dim3(S_, H_), 256, 0, stream>>>(sc, tokb);
      // head-averaged attention output: tmp = (1/H) * sum_h P_h V_h
      k_zero<<<dim3(S_ * D_ / 256), 256, 0, stream>>>(tmp);
      gemm_wmma<128, 128, 2, 4, false><<<dim3(D_ / 128, S_ / 128, H_), 256, 0, stream>>>(
          sc, (long)S_ * S_, S_,
          v, (long)S_ * D_, D_,
          tmp, 0L, D_,
          nullptr, 0L,
          S_, 1.0f / (float)H_, 0, 1);
      // x = LN(x + attn_out)
      k_add_ln<<<dim3(S_), 256, 0, stream>>>(xb, tmp, g1 + l * D_, be1 + l * D_);
      // ffh = relu(x W1 + b1)
      gemm_wmma<128, 128, 2, 4, false><<<dim3(FF_ / 128, S_ / 128, 1), 256, 0, stream>>>(
          xb, 0L, D_,
          W1 + (long)l * D_ * FF_, 0L, FF_,
          ffh, 0L, FF_,
          b1 + (long)l * FF_, 0L,
          D_, 1.0f, 1, 0);
      // tmp = ffh W2 + b2
      gemm_wmma<128, 128, 2, 4, false><<<dim3(D_ / 128, S_ / 128, 1), 256, 0, stream>>>(
          ffh, 0L, FF_,
          W2 + (long)l * FF_ * D_, 0L, D_,
          tmp, 0L, D_,
          b2 + (long)l * D_, 0L,
          FF_, 1.0f, 0, 0);
      // x = LN(x + ff)
      k_add_ln<<<dim3(S_), 256, 0, stream>>>(xb, tmp, g2 + l * D_, be2 + l * D_);
    }
  }

  // logits = x Wout + bout   ([2048,768] x [768,32000])
  gemm_wmma<128, 128, 2, 4, false><<<dim3(V_ / 128, (NB_ * S_) / 128, 1), 256, 0, stream>>>(
      x, 0L, D_,
      Wout, 0L, V_,
      out, 0L, V_,
      bout, 0L,
      D_, 1.0f, 0, 0);
}
